// Net_6107443494970
// MI455X (gfx1250) — compile-verified, matrix operand order checked by reading, not run on hardware
//
#include <hip/hip_runtime.h>
#include <hip/hip_bf16.h>

typedef __attribute__((ext_vector_type(16))) __bf16 v16bf;
typedef __attribute__((ext_vector_type(8)))  float  v8f;

// ---------- helpers ----------
static __device__ __forceinline__ __bf16 tobf(float f) { return (__bf16)f; }

static __device__ __forceinline__ unsigned fkey(float f) {
  unsigned b = __float_as_uint(f);
  return (b & 0x80000000u) ? ~b : (b | 0x80000000u); // monotone float->uint
}
static __device__ __forceinline__ float funkey(unsigned k) {
  unsigned b = (k & 0x80000000u) ? (k & 0x7fffffffu) : ~k;
  return __uint_as_float(b);
}

// ---------- fills ----------
__global__ void fill_f32(float* __restrict__ p, float v, long long n) {
  long long i = (long long)blockIdx.x * blockDim.x + threadIdx.x;
  if (i < n) p[i] = v;
}
__global__ void fill_u32(unsigned* __restrict__ p, unsigned v, long long n) {
  long long i = (long long)blockIdx.x * blockDim.x + threadIdx.x;
  if (i < n) p[i] = v;
}

// ---------- edge scatter: msg[dst] += h[src]; cnt[dst] += 1 ----------
__global__ void edge_scatter(const float* __restrict__ h,
                             const int* __restrict__ src,
                             const int* __restrict__ dst,
                             float* __restrict__ msg,
                             float* __restrict__ cnt, long long E) {
  long long idx = (long long)blockIdx.x * blockDim.x + threadIdx.x;
  long long e = idx >> 6;
  int f = (int)(idx & 63);
  if (e >= E) return;
  if (f == 0) {
    __builtin_prefetch(src + e + 2048, 0, 0);   // global_prefetch_b8
    __builtin_prefetch(dst + e + 2048, 0, 0);
  }
  int s = src[e];
  int d = dst[e];
  atomicAdd(&msg[(long long)d * 64 + f], h[(long long)s * 64 + f]);
  if (f == 0) atomicAdd(&cnt[d], 1.0f);
}

// ---------- mean = msg / cnt (in place) ----------
__global__ void mean_finalize(float* __restrict__ msg, const float* __restrict__ cnt,
                              long long n64) {
  long long idx = (long long)blockIdx.x * blockDim.x + threadIdx.x;
  if (idx >= n64) return;
  float c = cnt[idx >> 6];
  msg[idx] = (c > 0.0f) ? msg[idx] / c : 0.0f;
}

// ---------- A fragment: 4x global_load_b128, hw bf16 convert ----------
// 16-bit A layout (ISA 7.12.2): lane l15 = row M, halves hold K 0..7 / 16..23
// (+8 for lanes 16..31) => two contiguous 8-float runs per lane.
static __device__ __forceinline__ v16bf load_a_frag(const float* __restrict__ A,
                                                    int row_base, int kbase,
                                                    int nrows, int l15, int half) {
  union { v16bf v; __bf16 e[16]; } u;
  int m = row_base + l15;
  if (m < nrows) {
    const float4* ap = (const float4*)(A + (long long)m * 64 + kbase + 8 * half);
    float4 q0 = ap[0];   // k+0..3
    float4 q1 = ap[1];   // k+4..7
    float4 q2 = ap[4];   // k+16..19
    float4 q3 = ap[5];   // k+20..23
    u.e[0]  = tobf(q0.x); u.e[1]  = tobf(q0.y); u.e[2]  = tobf(q0.z); u.e[3]  = tobf(q0.w);
    u.e[4]  = tobf(q1.x); u.e[5]  = tobf(q1.y); u.e[6]  = tobf(q1.z); u.e[7]  = tobf(q1.w);
    u.e[8]  = tobf(q2.x); u.e[9]  = tobf(q2.y); u.e[10] = tobf(q2.z); u.e[11] = tobf(q2.w);
    u.e[12] = tobf(q3.x); u.e[13] = tobf(q3.y); u.e[14] = tobf(q3.z); u.e[15] = tobf(q3.w);
  } else {
#pragma unroll
    for (int j = 0; j < 16; ++j) u.e[j] = tobf(0.0f);
  }
  return u.v;
}

// ---------- B fragment: 2x ds_load_b128 from pre-swizzled LDS ----------
static __device__ __forceinline__ v16bf load_b_frag(const unsigned short* __restrict__ Bs,
                                                    int frag, int lane) {
  union { v16bf v; uint4 q[2]; } u;
  const uint4* p = (const uint4*)(Bs + ((size_t)(frag * 32 + lane) << 4));
  u.q[0] = p[0];
  u.q[1] = p[1];
  return u.v;
}

// ---------- stage weights into LDS in B-fragment order ----------
// frag fi = kt*4 + nt; element j of lane l: k = kt*32 + 8*(l>>4) + off(j) ; n = nt*16 + (l&15)
static __device__ __forceinline__ void stage_weights(const float* __restrict__ W,
                                                     unsigned short* __restrict__ Bs,
                                                     int tid) {
  for (int i = tid; i < 8 * 32 * 16; i += 128) {
    int fi   = i >> 9;
    int lane = (i >> 4) & 31;
    int j    = i & 15;
    int kt = fi >> 2, nt = fi & 3;
    int pj = j >> 1;
    int off = ((pj < 4) ? (2 * pj) : (2 * pj + 8)) + (j & 1);
    int k = kt * 32 + 8 * (lane >> 4) + off;
    int n = nt * 16 + (lane & 15);
    union { __bf16 b; unsigned short s; } c;
    c.b = tobf(W[k * 64 + n]);
    Bs[i] = c.s;
  }
}

// ---------- fused GEMM: out = A1@W1 [+ A2@W2] + bias, optional relu ----------
// blockDim = 128 (4 waves); persistent loop over 16-row tiles; each tile does
// 2 kt x 4 nt wmma per input matrix.
__global__ __launch_bounds__(128)
void gemm64_wmma(const float* __restrict__ A1, const float* __restrict__ A2,
                 const float* __restrict__ W1, const float* __restrict__ W2,
                 const float* __restrict__ bias, float* __restrict__ out,
                 int nrows, int relu, int use2) {
  __shared__ __align__(16) unsigned short b1s[8 * 32 * 16];
  __shared__ __align__(16) unsigned short b2s[8 * 32 * 16];
  int tid = threadIdx.x;
  stage_weights(W1, b1s, tid);
  if (use2) stage_weights(W2, b2s, tid);
  __syncthreads();

  int lane = tid & 31;
  int wave = tid >> 5;
  int l15 = lane & 15;
  int half = (lane >> 4) & 1;
  int ntiles = (nrows + 15) >> 4;

  float bvals[4];
#pragma unroll
  for (int nt = 0; nt < 4; ++nt) bvals[nt] = bias[nt * 16 + l15];

  for (int tile = blockIdx.x * 4 + wave; tile < ntiles; tile += gridDim.x * 4) {
    int row_base = tile << 4;

    v8f acc[4];
#pragma unroll
    for (int nt = 0; nt < 4; ++nt) {
#pragma unroll
      for (int r = 0; r < 8; ++r) acc[nt][r] = bvals[nt];
    }

#pragma unroll
    for (int kt = 0; kt < 2; ++kt) {
      v16bf a = load_a_frag(A1, row_base, kt * 32, nrows, l15, half);
#pragma unroll
      for (int nt = 0; nt < 4; ++nt) {
        v16bf b = load_b_frag(b1s, kt * 4 + nt, lane);
        acc[nt] = __builtin_amdgcn_wmma_f32_16x16x32_bf16(
            false, a, false, b, (short)0, acc[nt], false, false);
      }
    }
    if (use2) {
#pragma unroll
      for (int kt = 0; kt < 2; ++kt) {
        v16bf a = load_a_frag(A2, row_base, kt * 32, nrows, l15, half);
#pragma unroll
        for (int nt = 0; nt < 4; ++nt) {
          v16bf b = load_b_frag(b2s, kt * 4 + nt, lane);
          acc[nt] = __builtin_amdgcn_wmma_f32_16x16x32_bf16(
              false, a, false, b, (short)0, acc[nt], false, false);
        }
      }
    }

    // C/D layout: VGPR r -> lanes 0-15: (M=r, N=lane); lanes 16-31: (M=r+8, N=lane-16)
#pragma unroll
    for (int nt = 0; nt < 4; ++nt) {
#pragma unroll
      for (int r = 0; r < 8; ++r) {
        int m = row_base + r + 8 * half;
        if (m < nrows) {
          float v = acc[nt][r];
          if (relu) v = fmaxf(v, 0.0f);
          out[(long long)m * 64 + nt * 16 + l15] = v;
        }
      }
    }
  }
}

// ---------- gate = h . gate_w + gate_b ; segment_max via monotone-uint atomicMax -
__global__ void gate_kernel(const float* __restrict__ h, const int* __restrict__ batch,
                            const float* __restrict__ gw, const float* __restrict__ gb,
                            float* __restrict__ gate, unsigned* __restrict__ mkey,
                            long long n) {
  long long i = (long long)blockIdx.x * blockDim.x + threadIdx.x;
  if (i >= n) return;
  float s = gb[0];
#pragma unroll
  for (int f = 0; f < 64; ++f) s += h[i * 64 + f] * gw[f];
  gate[i] = s;
  atomicMax(&mkey[batch[i]], fkey(s));
}

// ---------- gate = exp(gate - max) ; den[b] += gate ----------
__global__ void exp_kernel(float* __restrict__ gate, const int* __restrict__ batch,
                           const unsigned* __restrict__ mkey, float* __restrict__ den,
                           long long n) {
  long long i = (long long)blockIdx.x * blockDim.x + threadIdx.x;
  if (i >= n) return;
  int b = batch[i];
  float ge = expf(gate[i] - funkey(mkey[b]));
  gate[i] = ge;
  atomicAdd(&den[b], ge);
}

// ---------- g[b] += (gate/den[b]) * h[i] ----------
__global__ void wsum_kernel(const float* __restrict__ h, const float* __restrict__ gate,
                            const int* __restrict__ batch, const float* __restrict__ den,
                            float* __restrict__ g, long long n) {
  long long idx = (long long)blockIdx.x * blockDim.x + threadIdx.x;
  long long i = idx >> 6;
  int f = (int)(idx & 63);
  if (i >= n) return;
  int b = batch[i];
  float alpha = gate[i] / den[b];
  atomicAdd(&g[(long long)b * 64 + f], alpha * h[i * 64 + f]);
}

// ---------- out = g1 . lin2_w + lin2_b ----------
__global__ void lin2_kernel(const float* __restrict__ g1, const float* __restrict__ w,
                            const float* __restrict__ b, float* __restrict__ out, int ng) {
  int i = blockIdx.x * blockDim.x + threadIdx.x;
  if (i >= ng) return;
  float s = b[0];
#pragma unroll
  for (int f = 0; f < 64; ++f) s += g1[(long long)i * 64 + f] * w[f];
  out[i] = s;
}

// ---------- host-side orchestration ----------
extern "C" void kernel_launch(void* const* d_in, const int* in_sizes, int n_in,
                              void* d_out, int out_size, void* d_ws, size_t ws_size,
                              hipStream_t stream) {
  const float* x   = (const float*)d_in[0];
  const int*  ei   = (const int*)d_in[1];
  const int*  batch= (const int*)d_in[2];
  const float* w1l = (const float*)d_in[3];
  const float* b1l = (const float*)d_in[4];
  const float* w1r = (const float*)d_in[5];
  const float* w2l = (const float*)d_in[6];
  const float* b2l = (const float*)d_in[7];
  const float* w2r = (const float*)d_in[8];
  const float* w3l = (const float*)d_in[9];
  const float* b3l = (const float*)d_in[10];
  const float* w3r = (const float*)d_in[11];
  const float* gw  = (const float*)d_in[12];
  const float* gb  = (const float*)d_in[13];
  const float* l1w = (const float*)d_in[14];
  const float* l1b = (const float*)d_in[15];
  const float* l2w = (const float*)d_in[16];
  const float* l2b = (const float*)d_in[17];

  long long N = in_sizes[0] / 64;
  long long E = in_sizes[1] / 2;
  int NG = out_size;
  const int* src = ei;
  const int* dst = ei + E;

  char* ws = (char*)d_ws;
  size_t off = 0;
  auto alloc = [&](size_t bytes) -> void* {
    void* p = ws + off;
    off = (off + bytes + 255) & ~(size_t)255;
    return p;
  };
  float*    h    = (float*)alloc((size_t)N * 64 * sizeof(float));
  float*    msg  = (float*)alloc((size_t)N * 64 * sizeof(float));
  float*    cnt  = (float*)alloc((size_t)N * sizeof(float));
  float*    gate = (float*)alloc((size_t)N * sizeof(float));
  unsigned* mkey = (unsigned*)alloc((size_t)NG * sizeof(unsigned));
  float*    den  = (float*)alloc((size_t)NG * sizeof(float));
  float*    g    = (float*)alloc((size_t)NG * 64 * sizeof(float));
  float*    g1   = (float*)alloc((size_t)NG * 64 * sizeof(float));

  dim3 blk(256);
  auto gridFor = [](long long n, int b) { return dim3((unsigned)((n + b - 1) / b)); };

  unsigned gemm_blocks_n = (unsigned)((N + 63) / 64);
  if (gemm_blocks_n > 512u) gemm_blocks_n = 512u;
  unsigned gemm_blocks_g = (unsigned)((NG + 63) / 64);
  if (gemm_blocks_g > 512u) gemm_blocks_g = 512u;

  const float* wls[3] = {w1l, w2l, w3l};
  const float* bls[3] = {b1l, b2l, b3l};
  const float* wrs[3] = {w1r, w2r, w3r};
  const float* cur = x;

  for (int L = 0; L < 3; ++L) {
    fill_f32<<<gridFor(N * 64, 256), blk, 0, stream>>>(msg, 0.0f, N * 64);
    fill_f32<<<gridFor(N, 256), blk, 0, stream>>>(cnt, 0.0f, N);
    edge_scatter<<<gridFor(E * 64, 256), blk, 0, stream>>>(cur, src, dst, msg, cnt, E);
    mean_finalize<<<gridFor(N * 64, 256), blk, 0, stream>>>(msg, cnt, N * 64);
    // h = relu(mean@wl + bl + cur@wr)   (in-place safe: row-block local)
    gemm64_wmma<<<dim3(gemm_blocks_n), dim3(128), 0, stream>>>(
        msg, cur, wls[L], wrs[L], bls[L], h, (int)N, /*relu=*/1, /*use2=*/1);
    cur = h;
  }

  // attentional pooling
  fill_u32<<<gridFor(NG, 256), blk, 0, stream>>>(mkey, 0u, NG);
  fill_f32<<<gridFor(NG, 256), blk, 0, stream>>>(den, 0.0f, NG);
  fill_f32<<<gridFor((long long)NG * 64, 256), blk, 0, stream>>>(g, 0.0f, (long long)NG * 64);
  gate_kernel<<<gridFor(N, 256), blk, 0, stream>>>(h, batch, gw, gb, gate, mkey, N);
  exp_kernel<<<gridFor(N, 256), blk, 0, stream>>>(gate, batch, mkey, den, N);
  wsum_kernel<<<gridFor(N * 64, 256), blk, 0, stream>>>(h, gate, batch, den, g, N);

  // MLP head: g1 = relu(g@lin1_w + lin1_b); out = g1@lin2_w + lin2_b
  gemm64_wmma<<<dim3(gemm_blocks_g), dim3(128), 0, stream>>>(
      g, (const float*)nullptr, l1w, l1w, l1b, g1, NG, /*relu=*/1, /*use2=*/0);
  lin2_kernel<<<gridFor(NG, 256), blk, 0, stream>>>(g1, l2w, l2b, (float*)d_out, NG);
}
